// BoundaryLoss_71433896067651
// MI455X (gfx1250) — compile-verified, matrix operand order checked by reading, not run on hardware
//
#include <hip/hip_runtime.h>

// Problem constants (N=16, C=4, H=256, W=256)
#define PN 16
#define PC 4
#define PH 256
#define PW 256
#define PHW (PH * PW)              // 65536
#define PCHW (PC * PHW)            // 262144
#define PTOT (PN * PCHW)           // 4194304
#define PPOS (PN * PHW)            // 1048576 spatial positions (n,h,w)
#define EDT_INF 1e10f

typedef __attribute__((ext_vector_type(16))) _Float16 v16h;
typedef __attribute__((ext_vector_type(8)))  float    v8f;

// Async global->LDS copy of 16 bytes (CDNA5 Tensor/async path, ASYNCcnt-tracked).
// lds_off: workgroup-relative LDS byte offset; gaddr: 64-bit global byte address.
__device__ __forceinline__ void async_copy_b128(unsigned lds_off, unsigned long long gaddr) {
  asm volatile("global_load_async_to_lds_b128 %0, %1, off"
               :: "v"(lds_off), "v"(gaddr) : "memory");
}
__device__ __forceinline__ void async_wait0() {
  asm volatile("s_wait_asynccnt 0x0" ::: "memory");
}

// ---------------------------------------------------------------------------
// K1: f = (target==0) ? INF : 0, then exact min-plus along C (L=4), in registers.
// ---------------------------------------------------------------------------
__global__ void k_init_cpass(const int* __restrict__ tgt, float* __restrict__ f) {
  int t = blockIdx.x * blockDim.x + threadIdx.x;
  if (t >= PPOS) return;
  int n = t >> 16;           // / PHW
  int rem = t & (PHW - 1);
  int base = n * PCHW + rem;

  float v[PC];
#pragma unroll
  for (int c = 0; c < PC; ++c)
    v[c] = (tgt[base + c * PHW] == 0) ? EDT_INF : 0.0f;

#pragma unroll
  for (int i = 0; i < PC; ++i) {
    float m = v[0] + (float)(i * i);
#pragma unroll
    for (int j = 1; j < PC; ++j) {
      int d = i - j;
      m = fminf(m, v[j] + (float)(d * d));
    }
    f[base + i * PHW] = m;
  }
}

// ---------------------------------------------------------------------------
// K2: min-plus along H. One block = full-height 256(H) x 16(W) tile staged
// into LDS with async global->LDS B128 DMA (no VGPR staging), computed as
// h^2 - 2*h*j + (f[j] + j^2) (1 FMA + 1 MIN per term), written back in place.
// Grid = N*C*(W/16) = 1024 blocks.
// ---------------------------------------------------------------------------
__global__ void k_edt_h(float* __restrict__ f) {
  __shared__ __attribute__((aligned(16))) float g[PH * 16];   // raw f tile [j][wl]
  int b  = blockIdx.x;
  int nc = b >> 4;                          // (n*C + c)
  int w0 = (b & 15) << 4;
  int base = nc * PHW + w0;
  int t  = threadIdx.x;

  // --- async stage: 4 x B128 per thread (256 rows x 16 floats) ---
  {
    int q  = (t & 3) << 2;                  // 16B chunk within a 16-float row
    int jj = t >> 2;                        // 0..63
#pragma unroll
    for (int r = 0; r < 4; ++r) {
      int j = (r << 6) + jj;
      unsigned lds = (unsigned)(uintptr_t)(&g[j * 16 + q]);
      unsigned long long ga = (unsigned long long)(uintptr_t)(f + base + j * PW + q);
      async_copy_b128(lds, ga);
    }
    async_wait0();
  }
  __syncthreads();

  int wl = t & 15;
  int hb = t >> 4;                          // 0..15

  float acc[16];
#pragma unroll
  for (int r = 0; r < 16; ++r) acc[r] = 3.4e38f;

  float hbF = (float)hb;
  for (int j = 0; j < PH; ++j) {
    float gv  = g[j * 16 + wl] + (float)(j * j);
    float m2j = -2.0f * (float)j;
#pragma unroll
    for (int r = 0; r < 16; ++r) {
      float h = hbF + (float)(r << 4);
      acc[r] = fminf(acc[r], fmaf(m2j, h, gv));
    }
  }

#pragma unroll
  for (int r = 0; r < 16; ++r) {
    int h = hb + (r << 4);
    f[base + h * PW + wl] = acc[r] + (float)(h * h);
  }
}

// ---------------------------------------------------------------------------
// K3: min-plus along W, fused with sqrt -> dist. One block = 16 rows x 256 w,
// async-staged into LDS (row stride 260 floats: keeps B128 16-B alignment and
// spreads rows across banks: 4r mod 64 distinct for r<16), in place.
// Grid = N*C*H/16 = 1024 blocks.
// ---------------------------------------------------------------------------
__global__ void k_edt_w_sqrt(float* __restrict__ f) {
  __shared__ __attribute__((aligned(16))) float g[16 * 260];
  int base = blockIdx.x * (16 * PW);
  int t = threadIdx.x;

  // --- async stage: 4 x B128 per thread (16 rows x 256 floats) ---
#pragma unroll
  for (int r = 0; r < 4; ++r) {
    int i   = t + (r << 8);                 // 0..1023 b128 chunks
    int row = i >> 6;                       // 64 chunks per row
    int c4  = (i & 63) << 2;                // float column
    unsigned lds = (unsigned)(uintptr_t)(&g[row * 260 + c4]);
    unsigned long long ga = (unsigned long long)(uintptr_t)(f + base + row * PW + c4);
    async_copy_b128(lds, ga);
  }
  async_wait0();
  __syncthreads();

  int row = t >> 4;
  int wb  = t & 15;
  const float* grow = &g[row * 260];

  float acc[16];
#pragma unroll
  for (int r = 0; r < 16; ++r) acc[r] = 3.4e38f;

  float wbF = (float)wb;
  for (int j = 0; j < PW; ++j) {
    float gv  = grow[j] + (float)(j * j);
    float m2j = -2.0f * (float)j;
#pragma unroll
    for (int r = 0; r < 16; ++r) {
      float w = wbF + (float)(r << 4);
      acc[r] = fminf(acc[r], fmaf(m2j, w, gv));
    }
  }

#pragma unroll
  for (int r = 0; r < 16; ++r) {
    int w = wb + (r << 4);
    f[base + row * PW + w] = sqrtf(acc[r] + (float)(w * w));
  }
}

// ---------------------------------------------------------------------------
// K4: softmax over C + partial reduction of num = sum(p*dist), den = sum(dist).
// Wave-level reduction via v_wmma_f32_16x16x32_f16 with a ones B matrix:
// lane puts its partial in A slot [0] -> row (lane%16), K in {0,8}. With B=1s,
// D[m][n] = partial_m + partial_{m+16}; summing a lane's 8 C/D VGPRs gives the
// 8-row half-sum, one shfl_xor(16) completes the 32-lane total. f32 accumulate.
// ---------------------------------------------------------------------------
__global__ void k_loss_partial(const float* __restrict__ pred,
                               const float* __restrict__ dist,
                               float* __restrict__ partial) {
  float num = 0.0f, den = 0.0f;
  int stride = gridDim.x * blockDim.x;     // 1024*256 = 262144, PPOS = 4*stride
  for (int t = blockIdx.x * blockDim.x + threadIdx.x; t < PPOS; t += stride) {
    int n = t >> 16;
    int rem = t & (PHW - 1);
    int base = n * PCHW + rem;

    // prefetch next grid-stride chunk (speculative; dropped if OOB)
    {
      int t2 = t + stride;
      int base2 = (t2 >> 16) * PCHW + (t2 & (PHW - 1));
      __builtin_prefetch(pred + base2, 0, 0);
      __builtin_prefetch(dist + base2, 0, 0);
    }

    float x0 = pred[base + 0 * PHW], x1 = pred[base + 1 * PHW];
    float x2 = pred[base + 2 * PHW], x3 = pred[base + 3 * PHW];
    float d0 = dist[base + 0 * PHW], d1 = dist[base + 1 * PHW];
    float d2 = dist[base + 2 * PHW], d3 = dist[base + 3 * PHW];

    float m  = fmaxf(fmaxf(x0, x1), fmaxf(x2, x3));
    float e0 = __expf(x0 - m), e1 = __expf(x1 - m);
    float e2 = __expf(x2 - m), e3 = __expf(x3 - m);
    float s  = e0 + e1 + e2 + e3;

    num += (e0 * d0 + e1 * d1 + e2 * d2 + e3 * d3) / s;
    den += d0 + d1 + d2 + d3;
  }

  // --- wave32 WMMA reduction (EXEC is all ones here: no divergence) ---
  v16h ones, an, ad;
#pragma unroll
  for (int i = 0; i < 16; ++i) {
    ones[i] = (_Float16)1.0f;
    an[i] = (_Float16)0.0f;
    ad[i] = (_Float16)0.0f;
  }
  an[0] = (_Float16)num;
  ad[0] = (_Float16)den;

  v8f cn = {};
  v8f cd = {};
  cn = __builtin_amdgcn_wmma_f32_16x16x32_f16(false, an, false, ones,
                                              (short)0, cn, false, false);
  cd = __builtin_amdgcn_wmma_f32_16x16x32_f16(false, ad, false, ones,
                                              (short)0, cd, false, false);

  float sn = cn[0] + cn[1] + cn[2] + cn[3] + cn[4] + cn[5] + cn[6] + cn[7];
  float sd = cd[0] + cd[1] + cd[2] + cd[3] + cd[4] + cd[5] + cd[6] + cd[7];
  sn += __shfl_xor(sn, 16, 32);   // combine row halves 0-7 and 8-15
  sd += __shfl_xor(sd, 16, 32);

  __shared__ float wnum[8], wden[8];
  int lane = threadIdx.x & 31;
  int wave = threadIdx.x >> 5;
  if (lane == 0) { wnum[wave] = sn; wden[wave] = sd; }
  __syncthreads();
  if (threadIdx.x == 0) {
    float bn = 0.0f, bd = 0.0f;
#pragma unroll
    for (int wv = 0; wv < 8; ++wv) { bn += wnum[wv]; bd += wden[wv]; }
    partial[2 * blockIdx.x + 0] = bn;
    partial[2 * blockIdx.x + 1] = bd;
  }
}

// ---------------------------------------------------------------------------
// K5: single-block deterministic finalize over 1024 block partials.
// ---------------------------------------------------------------------------
__global__ void k_finalize(const float* __restrict__ partial, float* __restrict__ out) {
  __shared__ float sN[256], sD[256];
  int t = threadIdx.x;
  float n = 0.0f, d = 0.0f;
#pragma unroll
  for (int i = 0; i < 4; ++i) {
    int idx = t + 256 * i;
    n += partial[2 * idx + 0];
    d += partial[2 * idx + 1];
  }
  sN[t] = n; sD[t] = d;
  __syncthreads();
  for (int s = 128; s > 0; s >>= 1) {
    if (t < s) { sN[t] += sN[t + s]; sD[t] += sD[t + s]; }
    __syncthreads();
  }
  if (t == 0) out[0] = sN[0] / (sD[0] + 1e-10f);
}

// ---------------------------------------------------------------------------
extern "C" void kernel_launch(void* const* d_in, const int* in_sizes, int n_in,
                              void* d_out, int out_size, void* d_ws, size_t ws_size,
                              hipStream_t stream) {
  const float* pred = (const float*)d_in[0];
  const int*   tgt  = (const int*)d_in[1];
  float* f       = (float*)d_ws;           // 4,194,304 floats (16 MB)
  float* partial = f + PTOT;               // 2048 floats
  float* out     = (float*)d_out;

  k_init_cpass <<<PPOS / 256, 256, 0, stream>>>(tgt, f);
  k_edt_h      <<<PN * PC * (PW / 16), 256, 0, stream>>>(f);   // 1024 blocks
  k_edt_w_sqrt <<<PN * PC * PH / 16, 256, 0, stream>>>(f);     // 1024 blocks
  k_loss_partial<<<1024, 256, 0, stream>>>(pred, f, partial);
  k_finalize   <<<1, 256, 0, stream>>>(partial, out);
}
